// AstrocyteGate_76141180223629
// MI455X (gfx1250) — matrix operand order repaired and checked by reading, not compiled, not run on hardware
//
#include <hip/hip_runtime.h>
#include <math.h>

// Problem constants (reference: B=8, T=2048, D=2048)
#define DDIM   2048
#define ROWS   16384           // B*T
#define NB1    256             // partial-sum blocks (deterministic 2-phase reduce)
#define SCALE  256.0f          // keep f16 WMMA inputs in normal range
#define ISCALE (1.0f / 256.0f)

static constexpr float kDecay = 0.99900049983337491f;   // exp(-1/1000)
static constexpr float kOneMinusDecay = 1.0f - 0.99900049983337491f;

typedef __attribute__((ext_vector_type(16))) _Float16 v16h;
typedef __attribute__((ext_vector_type(8)))  float    v8f;

// ---------------------------------------------------------------------------
// Kernel 1: per-block partial column sums of x  (deterministic, coalesced b128)
// grid = NB1 blocks x 256 threads; thread t owns float4 columns t and t+256.
// ---------------------------------------------------------------------------
__global__ void k_pool_partial(const float* __restrict__ x,
                               float* __restrict__ partials) {
    const int t = threadIdx.x;
    float4 a0 = make_float4(0.f, 0.f, 0.f, 0.f);
    float4 a1 = make_float4(0.f, 0.f, 0.f, 0.f);
    for (int r = blockIdx.x; r < ROWS; r += gridDim.x) {
        const float4* row = (const float4*)(x + (size_t)r * DDIM);
        float4 v0 = row[t];
        float4 v1 = row[t + 256];
        a0.x += v0.x; a0.y += v0.y; a0.z += v0.z; a0.w += v0.w;
        a1.x += v1.x; a1.y += v1.y; a1.z += v1.z; a1.w += v1.w;
    }
    float4* p = (float4*)(partials + (size_t)blockIdx.x * DDIM);
    p[t] = a0;
    p[t + 256] = a1;
}

// ---------------------------------------------------------------------------
// Kernel 2: reduce NB1 partials -> pooled, pre-scaled by SCALE for f16 WMMA.
// grid = 8 blocks x 256 threads (one thread per column).
// ---------------------------------------------------------------------------
__global__ void k_pool_reduce(const float* __restrict__ partials,
                              float* __restrict__ pooled_scaled) {
    const int d = blockIdx.x * 256 + threadIdx.x;
    float s = 0.f;
    for (int b = 0; b < NB1; ++b) s += partials[(size_t)b * DDIM + d];
    pooled_scaled[d] = s * (SCALE / (float)ROWS);
}

// ---------------------------------------------------------------------------
// WMMA GEMV tile: rows [mbase, mbase+16) of W (f32 row-major, 2048 cols) times
// vec[2048] (f32, pre-scaled). B columns all carry vec -> D column 0 = W@vec.
// One full wave32, EXEC all ones. Emits 64x v_wmma_f32_16x16x32_f16.
// ---------------------------------------------------------------------------
__device__ __forceinline__ v8f wmma_gemv_tile(const float* __restrict__ W,
                                              const float* __restrict__ vec,
                                              int mbase) {
    const int lane = threadIdx.x & 31;
    const int m    = lane & 15;
    const int kh8  = (lane >> 4) << 3;   // 0 or 8
    const int kh16 = kh8 << 1;           // 0 or 16
    const float* wrow = W + (size_t)(mbase + m) * DDIM;

    v8f c = {0.f, 0.f, 0.f, 0.f, 0.f, 0.f, 0.f, 0.f};
    for (int kb = 0; kb < DDIM; kb += 32) {
        const float* aLo = wrow + kb + kh8;        // K = kb+8*kh .. +7
        const float* aHi = wrow + kb + 16 + kh8;   // K = kb+16+8*kh .. +7
        const float* bp  = vec + kb + kh8;         // K = kb+8*kh .. +7, kb+16+8*kh .. +7
        v16h a, b;
#pragma unroll
        for (int j = 0; j < 8; ++j) {
            a[j]     = (_Float16)aLo[j];
            a[8 + j] = (_Float16)aHi[j];
        }
#pragma unroll
        for (int j = 0; j < 8; ++j) { b[j] = (_Float16)bp[j]; b[8 + j] = (_Float16)bp[16 + j]; }
        c = __builtin_amdgcn_wmma_f32_16x16x32_f16(false, a, false, b,
                                                   (short)0, c, false, false);
    }
    return c;
}

// ---------------------------------------------------------------------------
// Kernel 3: update = proj_w@pooled + proj_b; new_state = EMA; store *SCALE.
// grid = 128 blocks x 32 threads (one wave per 16-row tile).
// ---------------------------------------------------------------------------
__global__ void k_state(const float* __restrict__ proj_w,
                        const float* __restrict__ pooled_scaled,
                        const float* __restrict__ proj_b,
                        const float* __restrict__ state,
                        float* __restrict__ state_scaled) {
    const int mbase = blockIdx.x * 16;
    v8f c = wmma_gemv_tile(proj_w, pooled_scaled, mbase);
    const int lane = threadIdx.x;
    if (lane == 0 || lane == 16) {
        const int half = (lane == 16) ? 8 : 0;
#pragma unroll
        for (int r = 0; r < 8; ++r) {
            const int d = mbase + half + r;
            const float upd = c[r] * ISCALE + proj_b[d];
            const float ns  = kDecay * state[d] + kOneMinusDecay * upd;
            state_scaled[d] = ns * SCALE;
        }
    }
}

// ---------------------------------------------------------------------------
// Kernel 4: gain = sigmoid(gate_w@new_state + gate_b).
// ---------------------------------------------------------------------------
__global__ void k_gain(const float* __restrict__ gate_w,
                       const float* __restrict__ state_scaled,
                       const float* __restrict__ gate_b,
                       float* __restrict__ gain) {
    const int mbase = blockIdx.x * 16;
    v8f c = wmma_gemv_tile(gate_w, state_scaled, mbase);
    const int lane = threadIdx.x;
    if (lane == 0 || lane == 16) {
        const int half = (lane == 16) ? 8 : 0;
#pragma unroll
        for (int r = 0; r < 8; ++r) {
            const int d = mbase + half + r;
            const float logit = c[r] * ISCALE + gate_b[d];
            gain[d] = 1.0f / (1.0f + __expf(-logit));
        }
    }
}

// ---------------------------------------------------------------------------
// Kernel 5: out = x * gain (gain broadcast per column). float4 streaming;
// x is L2-resident (128 MiB < 192 MB) from pass 1.
// ---------------------------------------------------------------------------
__global__ void k_apply(const float* __restrict__ x,
                        const float* __restrict__ gain,
                        float* __restrict__ out) {
    const size_t n4 = (size_t)ROWS * (DDIM / 4);       // 8,388,608 float4
    const size_t stride = (size_t)gridDim.x * blockDim.x;
    const float4* x4 = (const float4*)x;
    const float4* g4 = (const float4*)gain;
    float4* o4 = (float4*)out;
    for (size_t i = (size_t)blockIdx.x * blockDim.x + threadIdx.x; i < n4;
         i += stride) {
        const float4 xv = x4[i];
        const float4 gv = g4[i & (DDIM / 4 - 1)];      // column float4 index
        float4 ov;
        ov.x = xv.x * gv.x;
        ov.y = xv.y * gv.y;
        ov.z = xv.z * gv.z;
        ov.w = xv.w * gv.w;
        o4[i] = ov;
    }
}

// ---------------------------------------------------------------------------
extern "C" void kernel_launch(void* const* d_in, const int* in_sizes, int n_in,
                              void* d_out, int out_size, void* d_ws, size_t ws_size,
                              hipStream_t stream) {
    const float* x      = (const float*)d_in[0];
    const float* state  = (const float*)d_in[1];
    const float* proj_w = (const float*)d_in[2];
    const float* proj_b = (const float*)d_in[3];
    const float* gate_w = (const float*)d_in[4];
    const float* gate_b = (const float*)d_in[5];
    float* out = (float*)d_out;

    float* ws            = (float*)d_ws;
    float* pooled_scaled = ws;                 // [2048]
    float* state_scaled  = ws + DDIM;          // [2048]
    float* gain          = ws + 2 * DDIM;      // [2048]
    float* partials      = ws + 4 * DDIM;      // [NB1 * 2048] = 2 MiB

    k_pool_partial<<<NB1, 256, 0, stream>>>(x, partials);
    k_pool_reduce<<<DDIM / 256, 256, 0, stream>>>(partials, pooled_scaled);
    k_state<<<DDIM / 16, 32, 0, stream>>>(proj_w, pooled_scaled, proj_b,
                                          state, state_scaled);
    k_gain<<<DDIM / 16, 32, 0, stream>>>(gate_w, state_scaled, gate_b, gain);
    k_apply<<<4096, 256, 0, stream>>>(x, gain, out);
}
